// APDL_RNN_11527692222527
// MI455X (gfx1250) — matrix operand order repaired and checked by reading, not compiled, not run on hardware
//
#include <hip/hip_runtime.h>
#include <hip/hip_bf16.h>
#include <math.h>

typedef __attribute__((ext_vector_type(16))) __bf16 v16bf;
typedef __attribute__((ext_vector_type(8)))  __bf16 v8bf;
typedef __attribute__((ext_vector_type(8)))  float  v8f;

#define TLEN 8192
#define HH   256
#define BSZ  32
// ws layout: [0, 16384): h exchange buffers, 2 groups x (16 x 256) bf16
//            [16384, 16448): barrier token slots, 2 groups x 8 ints
#define HX_BYTES   (2 * 16 * 256 * 2)
#define SLOT_OFF   HX_BYTES

__global__ __launch_bounds__(256, 1)
void apdl_lstm_scan(const float* __restrict__ x,
                    const float* __restrict__ W_ih,
                    const float* __restrict__ W_hh,
                    const float* __restrict__ b_ih,
                    const float* __restrict__ b_hh,
                    float* __restrict__ out,
                    __bf16* __restrict__ hx_all,
                    int* __restrict__ slots_all)
{
    const int tid  = threadIdx.x;
    const int wgid = blockIdx.x;
    const int grp  = wgid >> 3;      // batch group: rows [16*grp, 16*grp+16)
    const int sl   = wgid & 7;       // H-slice: indices [32*sl, 32*sl+32)
    const int lane = tid & 31;
    const int wave = tid >> 5;       // 8 waves; wave w owns N-tile w
    const int hi   = (lane >= 16) ? 1 : 0;
    const int ln   = lane & 15;

    __bf16* hx      = hx_all    + grp * (16 * 256);
    int*    myslots = slots_all + grp * 8;

    __shared__ float gl[16][132];        // gate staging (128 local cols, padded)
    __shared__ float sbuf[2][16][64];    // s ping-pong: [h(32) | c(32)] per row
    __shared__ float apb[16][64];        // all-pass state
    __shared__ float xs[16];             // x_t for this group's rows

    for (int i = tid; i < 16 * 64; i += 256) {
        int r = i >> 6, c = i & 63;
        sbuf[0][r][c] = 0.f; sbuf[1][r][c] = 0.f; apb[r][c] = 0.f;
    }
    __syncthreads();

    // Wave/lane constants: gate column for this lane's C/D slot
    const int q    = wave >> 1;              // gate quadrant (i,f,g,o)
    const int j0   = (wave & 1) * 16;
    const int gcol = q * 256 + sl * 32 + j0 + ln;
    const float wv    = W_ih[gcol];
    const float biasv = b_ih[gcol] + b_hh[gcol];

    // Preload persistent B fragments: B[k][n] = W_hh[gcol][k]  (8 K-tiles)
    v16bf Bfrag[8];
    {
        const float* wrow = W_hh + (size_t)gcol * HH;
        for (int kt = 0; kt < 8; ++kt) {
            const int kb = kt * 32 + hi * 8;
            v16bf f;
            #pragma unroll
            for (int e = 0; e < 16; ++e) f[e] = (__bf16)wrow[kb + (e < 8 ? e : e + 8)];
            Bfrag[kt] = f;
        }
    }

    const float coeff = (1.0f - 0.5f) / 1.5f;   // OS_FACTOR=1.5 -> 1/3

    for (int t = 0; t < TLEN; ++t) {
        const int cur = t & 1;       // s1 buffer
        const int oth = cur ^ 1;     // s2 buffer; also receives s_new

        // ---- Phase A: all-pass update, publish h_in (bf16) to L2 exchange
        if (tid < 16) xs[tid] = x[(size_t)(grp * 16 + tid) * TLEN + t];
        for (int p = tid; p < 16 * 32; p += 256) {
            int b = p >> 5, j = p & 31;
            float aph = coeff * (sbuf[cur][b][j]      - apb[b][j])      + sbuf[oth][b][j];
            float apc = coeff * (sbuf[cur][b][32 + j] - apb[b][32 + j]) + sbuf[oth][b][32 + j];
            apb[b][j]      = aph;
            apb[b][32 + j] = apc;
            hx[b * 256 + sl * 32 + j] = (__bf16)aph;
        }
        __syncthreads();   // all waves: s_wait_storecnt 0 -> h data at least in WGP$

        // ---- Inter-WG token barrier (8 WGs per group), agent scope.
        // Single release fence before publish; relaxed polls; single acquire
        // fence after all tokens observed (one WGP$-wide global_inv serves
        // every wave in this block before the trailing __syncthreads).
        if (tid == 0) {
            __builtin_amdgcn_fence(__ATOMIC_RELEASE, "agent");
            __hip_atomic_store(&myslots[sl], t + 1, __ATOMIC_RELAXED, __HIP_MEMORY_SCOPE_AGENT);
            for (int o = 0; o < 8; ++o)
                while (__hip_atomic_load(&myslots[o], __ATOMIC_RELAXED, __HIP_MEMORY_SCOPE_AGENT) < t + 1)
                    __builtin_amdgcn_s_sleep(1);
            __builtin_amdgcn_fence(__ATOMIC_ACQUIRE, "agent");
        }
        __syncthreads();

        // ---- Matmul: 16x16 gate tile per wave = h(16x256) @ Whh_slice^T
        v8f acc = {};
        #pragma unroll
        for (int kt = 0; kt < 8; ++kt) {
            const int kb = kt * 32 + hi * 8;
            const __bf16* arow = hx + ln * 256 + kb;
            v8bf lo  = *(const v8bf*)(arow);
            v8bf hi8 = *(const v8bf*)(arow + 16);
            v16bf a;
            #pragma unroll
            for (int e = 0; e < 8; ++e) { a[e] = lo[e]; a[8 + e] = hi8[e]; }
            acc = __builtin_amdgcn_wmma_f32_16x16x32_bf16(
                      false, a, false, Bfrag[kt], (short)0, acc, false, false);
        }

        // fold xg_t + biases, stage to LDS (C/D layout: VGPR r -> M = r + 8*hi)
        #pragma unroll
        for (int r = 0; r < 8; ++r) {
            int brow = r + hi * 8;
            gl[brow][q * 32 + j0 + ln] = acc[r] + xs[brow] * wv + biasv;
        }
        __syncthreads();

        // ---- Phase B: activations, state shift, stream outputs (NT stores)
        for (int p = tid; p < 16 * 32; p += 256) {
            int b = p >> 5, j = p & 31;
            float ig = gl[b][j];
            float fg = gl[b][32 + j];
            float gg = gl[b][64 + j];
            float og = gl[b][96 + j];
            float c  = apb[b][32 + j];
            float si = 1.f / (1.f + __expf(-ig));
            float sf = 1.f / (1.f + __expf(-fg));
            float so = 1.f / (1.f + __expf(-og));
            float cn = sf * c + si * tanhf(gg);
            float hn = so * tanhf(cn);
            sbuf[oth][b][j]      = hn;   // s_new overwrites consumed s2 buffer
            sbuf[oth][b][32 + j] = cn;
            int bg = grp * 16 + b;
            size_t o1 = ((size_t)bg * TLEN + t) * HH + (sl * 32 + j);
            const size_t plane = (size_t)BSZ * TLEN * HH;
            __builtin_nontemporal_store(hn, &out[o1]);             // y
            __builtin_nontemporal_store(hn, &out[plane + o1]);     // states[..., :H]
            __builtin_nontemporal_store(cn, &out[2 * plane + o1]); // states[..., H:]
        }
        __syncthreads();
    }

    // ap_final: (B, 2H)
    for (int p = tid; p < 16 * 32; p += 256) {
        int b = p >> 5, j = p & 31;
        int bg = grp * 16 + b;
        const size_t base = 3 * (size_t)BSZ * TLEN * HH;
        out[base + (size_t)bg * 512 +       sl * 32 + j] = apb[b][j];
        out[base + (size_t)bg * 512 + 256 + sl * 32 + j] = apb[b][32 + j];
    }
}

extern "C" void kernel_launch(void* const* d_in, const int* in_sizes, int n_in,
                              void* d_out, int out_size, void* d_ws, size_t ws_size,
                              hipStream_t stream) {
    const float* x    = (const float*)d_in[0];
    const float* W_ih = (const float*)d_in[1];
    const float* W_hh = (const float*)d_in[2];
    const float* b_ih = (const float*)d_in[3];
    const float* b_hh = (const float*)d_in[4];
    float* out = (float*)d_out;

    __bf16* hx    = (__bf16*)d_ws;
    int*    slots = (int*)((char*)d_ws + SLOT_OFF);

    // Reset barrier tokens every launch (capture-legal memset node).
    hipMemsetAsync((char*)d_ws + SLOT_OFF, 0, 2 * 8 * sizeof(int), stream);

    apdl_lstm_scan<<<16, 256, 0, stream>>>(x, W_ih, W_hh, b_ih, b_hh, out, hx, slots);
}
